// slot_dec_27728308863822
// MI455X (gfx1250) — compile-verified, matrix operand order checked by reading, not run on hardware
//
#include <hip/hip_runtime.h>

typedef unsigned short u16;
typedef __attribute__((ext_vector_type(16))) __bf16 v16bf;
typedef __attribute__((ext_vector_type(8)))  float  v8f;

#define H      1024
#define L      512
#define IN4    4096   // width of f_t = [x_t, hi_t] (2H + 2H)
#define K5     5120   // W_ih inner dim
#define GATES  4096   // 4H
#define LAB    122
#define KC     128    // IN4 / 32 K-chunks for wmma
#define NBLK   16     // workgroups in persistent recurrent kernel

// ---------- helpers ----------

static __device__ inline u16 f2bf(float f) {
    union { float f; unsigned u; } v; v.f = f;
    unsigned r = v.u + 0x7FFFu + ((v.u >> 16) & 1u);   // round-to-nearest-even
    return (u16)(r >> 16);
}

static __device__ inline float sigmoidf(float x) { return 1.0f / (1.0f + __expf(-x)); }

static __device__ inline v8f wmma_bf16(v16bf a, v16bf b, v8f c) {
    return __builtin_amdgcn_wmma_f32_16x16x32_bf16(false, a, false, b, (short)0, c,
                                                   false, false);
}

// fragment swizzle: kk (0..31) -> (lane group g, element e) per ISA 7.12.2
static __device__ inline void frag_ge(int kk, int& g, int& e) {
    g = (kk >> 3) & 1;
    e = (kk & 7) + ((kk >> 4) << 3);
}

// ---------- prep kernels ----------

// feats[t,k] = k<2048 ? x[t,k] : hi[t,k-2048], swizzled to A-fragment order (bf16)
__global__ void __launch_bounds__(256)
prep_feats(const float* __restrict__ x, const float* __restrict__ hi, u16* __restrict__ Asw) {
    int idx = blockIdx.x * 256 + threadIdx.x;          // 0 .. L*IN4-1
    int t = idx >> 12, k = idx & 4095;
    float val = (k < 2048) ? x[t * 2048 + k] : hi[t * 2048 + (k - 2048)];
    int mT = t >> 4, m = t & 15;
    int kc = k >> 5, kk = k & 31;
    int g, e; frag_ge(kk, g, e);
    int lane = m + (g << 4);
    Asw[((size_t)(mT * KC + kc) * 32 + lane) * 16 + e] = f2bf(val);
}

// W_x[r,k] = W_ih[r, k] (k < 4096), swizzled to B-fragment order (bf16)
__global__ void __launch_bounds__(256)
prep_wx(const float* __restrict__ W_ih, u16* __restrict__ Bsw) {
    int idx = blockIdx.x * 256 + threadIdx.x;          // 0 .. GATES*IN4-1
    int r = idx >> 12, k = idx & 4095;
    int nT = r >> 4, n = r & 15;
    int kc = k >> 5, kk = k & 31;
    int g, e; frag_ge(kk, g, e);
    int lane = n + (g << 4);
    Bsw[((size_t)(nT * KC + kc) * 32 + lane) * 16 + e] = f2bf(W_ih[(size_t)r * K5 + k]);
}

// Wr[r,j] = W_ih[r, 4096+j] + W_hh[r,j]   (fp32, row-major)
__global__ void __launch_bounds__(256)
prep_wr(const float* __restrict__ W_ih, const float* __restrict__ W_hh, float* __restrict__ Wr) {
    int idx = blockIdx.x * 256 + threadIdx.x;          // 0 .. GATES*H-1
    int r = idx >> 10, j = idx & 1023;
    Wr[(size_t)r * H + j] = W_ih[(size_t)r * K5 + IN4 + j] + W_hh[(size_t)r * H + j];
}

__global__ void __launch_bounds__(256)
prep_bias(const float* __restrict__ b_ih, const float* __restrict__ b_hh, float* __restrict__ bias) {
    int i = blockIdx.x * 256 + threadIdx.x;
    if (i < GATES) bias[i] = b_ih[i] + b_hh[i];
}

// ---------- big input GEMM: G[t,r] = sum_k feats[t,k] * W_ih[r,k] ----------
// wave -> 32x64 strip: (2 consecutive mT) x (4 consecutive nT).
// 16 mT-pairs x 64 nQuads = 1024 waves = 128 blocks of 8 waves.
// Double-buffered: prefetch chunk kc+1 while the 8 WMMAs of chunk kc execute.
__global__ void __launch_bounds__(256)
wmma_gemm_inp(const u16* __restrict__ Asw, const u16* __restrict__ Bsw, float* __restrict__ G) {
    const int wave = (blockIdx.x * 256 + threadIdx.x) >> 5;
    const int lane = threadIdx.x & 31;
    const int mT0  = (wave >> 6) * 2;  // 0,2,..,30
    const int nT0  = (wave & 63) * 4;  // 0..252

    v8f c00 = {}, c01 = {}, c02 = {}, c03 = {};
    v8f c10 = {}, c11 = {}, c12 = {}, c13 = {};

    const u16* aB0 = Asw + (size_t)(mT0 + 0) * KC * 512 + lane * 16;
    const u16* aB1 = Asw + (size_t)(mT0 + 1) * KC * 512 + lane * 16;
    const u16* pb0 = Bsw + (size_t)(nT0 + 0) * KC * 512 + lane * 16;
    const u16* pb1 = Bsw + (size_t)(nT0 + 1) * KC * 512 + lane * 16;
    const u16* pb2 = Bsw + (size_t)(nT0 + 2) * KC * 512 + lane * 16;
    const u16* pb3 = Bsw + (size_t)(nT0 + 3) * KC * 512 + lane * 16;

#define LOADF(p, kc) (*(const v16bf*)((p) + (size_t)(kc) * 512))

    v16bf a0 = LOADF(aB0, 0), a1 = LOADF(aB1, 0);
    v16bf b0 = LOADF(pb0, 0), b1 = LOADF(pb1, 0);
    v16bf b2 = LOADF(pb2, 0), b3 = LOADF(pb3, 0);

#pragma unroll 1
    for (int kc = 0; kc < KC - 1; ++kc) {
        // prefetch next chunk into fresh registers (no dep on current WMMAs)
        v16bf na0 = LOADF(aB0, kc + 1), na1 = LOADF(aB1, kc + 1);
        v16bf nb0 = LOADF(pb0, kc + 1), nb1 = LOADF(pb1, kc + 1);
        v16bf nb2 = LOADF(pb2, kc + 1), nb3 = LOADF(pb3, kc + 1);

        c00 = wmma_bf16(a0, b0, c00);  c10 = wmma_bf16(a1, b0, c10);
        c01 = wmma_bf16(a0, b1, c01);  c11 = wmma_bf16(a1, b1, c11);
        c02 = wmma_bf16(a0, b2, c02);  c12 = wmma_bf16(a1, b2, c12);
        c03 = wmma_bf16(a0, b3, c03);  c13 = wmma_bf16(a1, b3, c13);

        a0 = na0; a1 = na1;
        b0 = nb0; b1 = nb1; b2 = nb2; b3 = nb3;
    }
    // tail chunk
    c00 = wmma_bf16(a0, b0, c00);  c10 = wmma_bf16(a1, b0, c10);
    c01 = wmma_bf16(a0, b1, c01);  c11 = wmma_bf16(a1, b1, c11);
    c02 = wmma_bf16(a0, b2, c02);  c12 = wmma_bf16(a1, b2, c12);
    c03 = wmma_bf16(a0, b3, c03);  c13 = wmma_bf16(a1, b3, c13);
#undef LOADF

    // C/D layout: vgpr v, lane -> row = v + 8*(lane>>4), col = lane&15
    const int g = lane >> 4, n = lane & 15;
    float* out0 = G + (size_t)((mT0 + 0) * 16 + g * 8) * GATES + nT0 * 16 + n;
    float* out1 = G + (size_t)((mT0 + 1) * 16 + g * 8) * GATES + nT0 * 16 + n;
#pragma unroll
    for (int v = 0; v < 8; ++v) {
        out0[(size_t)v * GATES +  0] = c00[v];
        out0[(size_t)v * GATES + 16] = c01[v];
        out0[(size_t)v * GATES + 32] = c02[v];
        out0[(size_t)v * GATES + 48] = c03[v];
        out1[(size_t)v * GATES +  0] = c10[v];
        out1[(size_t)v * GATES + 16] = c11[v];
        out1[(size_t)v * GATES + 32] = c12[v];
        out1[(size_t)v * GATES + 48] = c13[v];
    }
}

// ---------- persistent recurrent kernel ----------

static __device__ inline void grid_barrier(unsigned* bar, unsigned& gen, unsigned nblk) {
    __threadfence();          // make this thread's global stores visible device-wide
    __syncthreads();
    ++gen;
    if (threadIdx.x == 0) {
        __hip_atomic_fetch_add(bar, 1u, __ATOMIC_RELEASE, __HIP_MEMORY_SCOPE_AGENT);
        while (__hip_atomic_load(bar, __ATOMIC_ACQUIRE, __HIP_MEMORY_SCOPE_AGENT) < gen * nblk)
            __builtin_amdgcn_s_sleep(1);
    }
    __syncthreads();
    __threadfence();          // acquire: discard stale cached lines
}

// 16 blocks x 256 threads = 4096 threads; thread r owns gate row r.
__global__ void __launch_bounds__(256)
lstm_recurrent(const float* __restrict__ G, const float* __restrict__ Wr,
               const float* __restrict__ bias, float* h, float* gates,
               float* __restrict__ outs, unsigned* bar) {
    __shared__ float sh[H];
    const int r = blockIdx.x * 256 + threadIdx.x;   // 0..4095
    const float brow = bias[r];
    const float4* __restrict__ w4 = (const float4*)(Wr + (size_t)r * H);
    float c = 0.0f;
    unsigned gen = 0;

    for (int t = 0; t < L; ++t) {
        // stage h into LDS (volatile: written by other WGPs last step)
        const volatile float* hv = h;
        for (int j = threadIdx.x; j < H; j += 256) sh[j] = hv[j];
        __syncthreads();

        float acc = G[(size_t)t * GATES + r] + brow;
        const float4* s4 = (const float4*)sh;
#pragma unroll 4
        for (int k = 0; k < H / 4; ++k) {
            float4 w = w4[k];
            float4 s = s4[k];
            acc += w.x * s.x + w.y * s.y + w.z * s.z + w.w * s.w;
        }
        gates[r] = acc;

        grid_barrier(bar, gen, NBLK);   // all 4096 gate values ready

        if (r < H) {
            const volatile float* gv = gates;
            float gi = gv[r], gf = gv[r + H], gg = gv[r + 2 * H], go = gv[r + 3 * H];
            c = sigmoidf(gf) * c + sigmoidf(gi) * tanhf(gg);
            float hn = sigmoidf(go) * tanhf(c);
            h[r] = hn;
            outs[(size_t)t * H + r] = hn;
        }

        grid_barrier(bar, gen, NBLK);   // h updated before anyone reloads it
    }
}

// ---------- final FC: out[t,l] = outs[t,:] . W_fc[l,:] + b_fc[l] ----------
__global__ void __launch_bounds__(128)
fc_kernel(const float* __restrict__ outs, const float* __restrict__ W_fc,
          const float* __restrict__ b_fc, float* __restrict__ out) {
    __shared__ float sh[H];
    const int t = blockIdx.x;
    for (int j = threadIdx.x; j < H; j += 128) sh[j] = outs[(size_t)t * H + j];
    __syncthreads();
    const int l = threadIdx.x;
    if (l < LAB) {
        float acc = b_fc[l];
        const float4* w4 = (const float4*)(W_fc + (size_t)l * H);
        const float4* s4 = (const float4*)sh;
#pragma unroll 4
        for (int k = 0; k < H / 4; ++k) {
            float4 w = w4[k];
            float4 s = s4[k];
            acc += w.x * s.x + w.y * s.y + w.z * s.z + w.w * s.w;
        }
        out[(size_t)t * LAB + l] = acc;
    }
}

// ---------- launcher ----------

extern "C" void kernel_launch(void* const* d_in, const int* in_sizes, int n_in,
                              void* d_out, int out_size, void* d_ws, size_t ws_size,
                              hipStream_t stream) {
    (void)in_sizes; (void)n_in; (void)out_size; (void)ws_size;
    const float* x    = (const float*)d_in[0];
    const float* hi   = (const float*)d_in[1];
    const float* W_ih = (const float*)d_in[2];
    const float* W_hh = (const float*)d_in[3];
    const float* b_ih = (const float*)d_in[4];
    const float* b_hh = (const float*)d_in[5];
    const float* W_fc = (const float*)d_in[6];
    const float* b_fc = (const float*)d_in[7];
    float* out = (float*)d_out;

    char* p = (char*)d_ws;
    u16*   Asw  = (u16*)p;       p += (size_t)L * IN4 * 2;          //  4 MB
    u16*   Bsw  = (u16*)p;       p += (size_t)GATES * IN4 * 2;      // 32 MB
    float* Wr   = (float*)p;     p += (size_t)GATES * H * 4;        // 16 MB
    float* G    = (float*)p;     p += (size_t)L * GATES * 4;        //  8 MB
    float* outs = (float*)p;     p += (size_t)L * H * 4;            //  2 MB
    float* bias = (float*)p;     p += (size_t)GATES * 4;
    float* h    = (float*)p;     p += (size_t)H * 4;
    float* gate = (float*)p;     p += (size_t)GATES * 4;
    unsigned* bar = (unsigned*)p;

    hipMemsetAsync(h,   0, H * sizeof(float), stream);
    hipMemsetAsync(bar, 0, 64, stream);

    prep_feats<<<(L * IN4) / 256, 256, 0, stream>>>(x, hi, Asw);
    prep_wx<<<(GATES * IN4) / 256, 256, 0, stream>>>(W_ih, Bsw);
    prep_wr<<<(GATES * H) / 256, 256, 0, stream>>>(W_ih, W_hh, Wr);
    prep_bias<<<GATES / 256, 256, 0, stream>>>(b_ih, b_hh, bias);

    wmma_gemm_inp<<<128, 256, 0, stream>>>(Asw, Bsw, G);

    lstm_recurrent<<<NBLK, 256, 0, stream>>>(G, Wr, bias, h, gate, outs, bar);

    fc_kernel<<<L, 128, 0, stream>>>(outs, W_fc, b_fc, out);
}
